// PointNetFeaturePropagation_26362509263295
// MI455X (gfx1250) — matrix operand order repaired and checked by reading, not compiled, not run on hardware
//
#include <hip/hip_runtime.h>
#include <hip/hip_bf16.h>
#include <math.h>

typedef __attribute__((ext_vector_type(16))) __bf16 v16bf;
typedef __attribute__((ext_vector_type(8)))  __bf16 v8bf;
typedef __attribute__((ext_vector_type(8)))  float  v8f;
typedef unsigned int u32x4 __attribute__((ext_vector_type(4)));
typedef int          i32x4 __attribute__((ext_vector_type(4)));
typedef int          i32x8 __attribute__((ext_vector_type(8)));

#define B_   8
#define N_   8192
#define S_   2048
#define C1_  128
#define C2_  256
#define CIN_ 384
#define H_   512
#define CHUNK_ROWS 256

// A-tile staging geometry: 128 rows x 32 k-elements per chunk, padded to 80B/row
// (64B data + 16B TDM pad) -> lane l reads at 80*l mod 256, conflict-free b128s.
#define AROW_ELEMS 40            // 80 bytes / 2
#define ABUF_BYTES (128 * 80)    // 10240 B per buffer

// ---------------- TDM: async 2D tile load Global->LDS (D# per CDNA5 ISA) ----------------
__device__ __forceinline__ void tdm_load_a_tile(unsigned ldsOff, const void* gsrc, int K_) {
    const unsigned long long ga = (unsigned long long)(uintptr_t)gsrc;
    u32x4 g0;
    g0[0] = 1u;                                                  // count=1, user mode
    g0[1] = ldsOff;                                              // lds_addr (bytes)
    g0[2] = (unsigned)ga;                                        // global_addr[31:0]
    g0[3] = (unsigned)((ga >> 32) & 0x01FFFFFFull) | (2u << 30); // addr[56:32] | type=2
    i32x8 g1;
    g1[0] = (1 << 16)      // data_size = 2 bytes
          | (1 << 20)      // pad_enable
          | (3 << 22)      // pad_interval: 16 DWORDs (= tile_dim0 row)
          | (3 << 25);     // pad_amount: 4 DWORDs (16 B)
    g1[1] = (int)((unsigned)(K_ & 0xFFFF) << 16);  // tensor_dim0[15:0]
    g1[2] = (K_ >> 16) & 0xFFFF;                   // tensor_dim0[31:16], tensor_dim1[15:0]=0
    g1[3] = 1 | (32 << 16);                        // tensor_dim1[31:16]=1 (64Ki rows), tile_dim0=32
    g1[4] = 128;                                   // tile_dim1=128 rows, tile_dim2=0
    g1[5] = K_;                                    // tensor_dim0_stride[31:0]
    g1[6] = 0;                                     // stride hi / dim1_stride
    g1[7] = 0;
    const i32x4 gz = {0, 0, 0, 0};
#if defined(__clang_major__) && (__clang_major__ >= 23)
    const i32x8 gz8 = {0, 0, 0, 0, 0, 0, 0, 0};
    __builtin_amdgcn_tensor_load_to_lds(g0, g1, gz, gz, gz8, 0);
#else
    __builtin_amdgcn_tensor_load_to_lds(g0, g1, gz, gz, 0);
#endif
}

// ---------------- 3-NN search: xyz2 (+|b|^2) staged in LDS, register top-3 ----------------
__global__ void knn3_kernel(const float* __restrict__ xyz1, const float* __restrict__ xyz2,
                            int* __restrict__ idx, float* __restrict__ wgt) {
    __shared__ float sx[S_], sy[S_], sz[S_], sr[S_];
    const int b = blockIdx.y;
    const float* p2 = xyz2 + (size_t)b * S_ * 3;
    for (int s = threadIdx.x; s < S_; s += blockDim.x) {
        const float bx = p2[s * 3 + 0], by = p2[s * 3 + 1], bz = p2[s * 3 + 2];
        sx[s] = bx; sy[s] = by; sz[s] = bz;
        sr[s] = bx * bx + by * by + bz * bz;
    }
    __syncthreads();
    const int n = blockIdx.x * blockDim.x + threadIdx.x;
    const size_t pt = (size_t)b * N_ + n;
    const float x = xyz1[pt * 3 + 0], y = xyz1[pt * 3 + 1], z = xyz1[pt * 3 + 2];
    const float ra = x * x + y * y + z * z;
    float d0 = 3.4e38f, d1 = 3.4e38f, d2 = 3.4e38f;
    int   i0 = 0, i1 = 0, i2 = 0;
    for (int s = 0; s < S_; ++s) {
        // |a|^2 - 2 a.b + |b|^2  (3 FMA + 1 FMA)
        float dot = x * sx[s];
        dot = fmaf(y, sy[s], dot);
        dot = fmaf(z, sz[s], dot);
        const float d = fmaf(-2.0f, dot, ra + sr[s]);
        if (d < d0)      { d2 = d1; i2 = i1; d1 = d0; i1 = i0; d0 = d; i0 = s; }
        else if (d < d1) { d2 = d1; i2 = i1; d1 = d;  i1 = s; }
        else if (d < d2) { d2 = d;  i2 = s; }
    }
    const float w0 = 1.0f / (fmaxf(d0, 0.0f) + 1e-8f);
    const float w1 = 1.0f / (fmaxf(d1, 0.0f) + 1e-8f);
    const float w2 = 1.0f / (fmaxf(d2, 0.0f) + 1e-8f);
    const float inv = 1.0f / (w0 + w1 + w2);
    idx[pt * 3 + 0] = i0; idx[pt * 3 + 1] = i1; idx[pt * 3 + 2] = i2;
    wgt[pt * 3 + 0] = w0 * inv; wgt[pt * 3 + 1] = w1 * inv; wgt[pt * 3 + 2] = w2 * inv;
}

// ---------------- interpolate + concat -> bf16 activation matrix (rows x 384) ----------------
__global__ void interp_concat_kernel(const float* __restrict__ points1, const float* __restrict__ points2,
                                     const int* __restrict__ idx, const float* __restrict__ wgt,
                                     __bf16* __restrict__ Xbf) {
    const size_t pt = blockIdx.x;
    const int b = (int)(pt >> 13);          // / N_
    const int c = threadIdx.x;              // 0..255
    const int*   id = idx + pt * 3;
    const float* w  = wgt + pt * 3;
    const float* p2 = points2 + (size_t)b * S_ * C2_;
    const float v = w[0] * p2[(size_t)id[0] * C2_ + c]
                  + w[1] * p2[(size_t)id[1] * C2_ + c]
                  + w[2] * p2[(size_t)id[2] * C2_ + c];
    __bf16* xr = Xbf + pt * CIN_;
    xr[C1_ + c] = (__bf16)v;
    if (c < C1_) xr[c] = (__bf16)points1[pt * C1_ + c];
}

// ---------------- fp32 -> bf16 cast (weights) ----------------
__global__ void f2bf_kernel(const float* __restrict__ in, __bf16* __restrict__ out, int n) {
    const int i = blockIdx.x * blockDim.x + threadIdx.x;
    if (i < n) out[i] = (__bf16)in[i];
}

// ---------------- bf16 WMMA GEMM with TDM-staged A tiles ----------------
// Block: 8 waves, 128 rows x 64 cols. A chunks (128x32) double-buffered in LDS via
// tensor_load_to_lds; B (weights) streamed from L2. Y = X*W^T + bias, fp32.
template <int K>
__global__ __launch_bounds__(256) void gemm_bf16_kernel(const __bf16* __restrict__ X,
                                                        const __bf16* __restrict__ W,
                                                        const float* __restrict__ bias,
                                                        float* __restrict__ Y) {
    __shared__ __bf16 bufA[2][128 * AROW_ELEMS];

    const int lane  = threadIdx.x & 31;
    const int wave  = threadIdx.x >> 5;
    const int m16   = lane & 15;
    const int khalf = lane >> 4;
    const int rowBase  = blockIdx.x * 128;
    const int colBase  = blockIdx.y * 64;
    const int rowLocal = wave * 16 + m16;

    const unsigned ldsBase = (unsigned)(uintptr_t)(&bufA[0][0]);

    // Prologue: kick off chunk 0, wait, publish.
    if (wave == 0) {
        tdm_load_a_tile(ldsBase, X + (size_t)rowBase * K, K);
        __builtin_amdgcn_s_wait_tensorcnt(0);
    }
    __syncthreads();

    v8f acc0 = {}, acc1 = {}, acc2 = {}, acc3 = {};
    const __bf16* wrow = W + (size_t)(colBase + m16) * K + khalf * 16;

    int cur = 0;
    for (int kb = 0; kb < K; kb += 32) {
        // Overlap: DMA the next A chunk into the other buffer while computing.
        if (wave == 0 && (kb + 32) < K) {
            tdm_load_a_tile(ldsBase + (unsigned)((cur ^ 1) * ABUF_BYTES),
                            X + (size_t)rowBase * K + kb + 32, K);
        }
        // A fragment from LDS (conflict-free: 80B row stride)
        const __bf16* rp = &bufA[cur][rowLocal * AROW_ELEMS];
        const v8bf lo = *(const v8bf*)(rp + khalf * 16);
        const v8bf hi = *(const v8bf*)(rp + khalf * 16 + 8);
        v16bf a;
#pragma unroll
        for (int i = 0; i < 8; ++i) { a[i] = lo[i]; a[i + 8] = hi[i]; }
        // B fragments: one output column's 16 contiguous K values per lane (L2-resident)
        const v16bf b0 = *(const v16bf*)(wrow + kb);
        const v16bf b1 = *(const v16bf*)(wrow + (size_t)16 * K + kb);
        const v16bf b2 = *(const v16bf*)(wrow + (size_t)32 * K + kb);
        const v16bf b3 = *(const v16bf*)(wrow + (size_t)48 * K + kb);
        acc0 = __builtin_amdgcn_wmma_f32_16x16x32_bf16(false, a, false, b0, (short)0, acc0, false, false);
        acc1 = __builtin_amdgcn_wmma_f32_16x16x32_bf16(false, a, false, b1, (short)0, acc1, false, false);
        acc2 = __builtin_amdgcn_wmma_f32_16x16x32_bf16(false, a, false, b2, (short)0, acc2, false, false);
        acc3 = __builtin_amdgcn_wmma_f32_16x16x32_bf16(false, a, false, b3, (short)0, acc3, false, false);

        __syncthreads();                               // everyone done with bufA[cur]
        if (wave == 0 && (kb + 32) < K) {
            __builtin_amdgcn_s_wait_tensorcnt(0);      // next chunk landed
        }
        __syncthreads();                               // publish next buffer
        cur ^= 1;
    }

    // Epilogue: C/D layout -> lane<16: row j, col lane; lane>=16: row j+8, col lane-16
    const int rowOut = rowBase + wave * 16 + khalf * 8;
#pragma unroll
    for (int t = 0; t < 4; ++t) {
        const v8f& acc = (t == 0) ? acc0 : (t == 1) ? acc1 : (t == 2) ? acc2 : acc3;
        const int col = colBase + t * 16 + m16;
        const float bv = bias[col];
#pragma unroll
        for (int j = 0; j < 8; ++j) {
            Y[(size_t)(rowOut + j) * H_ + col] = acc[j] + bv;
        }
    }
}

// ---------------- per-channel sum / sumsq over rows (coalesced, float atomics) ----------------
__global__ void zero_kernel(float* __restrict__ p, int n) {
    const int i = blockIdx.x * blockDim.x + threadIdx.x;
    if (i < n) p[i] = 0.0f;
}

__global__ void colstats_kernel(const float* __restrict__ Y, float* __restrict__ sum, float* __restrict__ sumsq) {
    const int t = threadIdx.x;                    // channels t and t+256
    const size_t r0 = (size_t)blockIdx.x * CHUNK_ROWS;
    float s0 = 0.f, q0 = 0.f, s1 = 0.f, q1 = 0.f;
    for (int r = 0; r < CHUNK_ROWS; ++r) {
        const float* row = Y + (r0 + r) * H_;
        const float a = row[t];
        const float b = row[t + 256];
        s0 += a; q0 += a * a;
        s1 += b; q1 += b * b;
    }
    atomicAdd(&sum[t], s0);         atomicAdd(&sumsq[t], q0);
    atomicAdd(&sum[t + 256], s1);   atomicAdd(&sumsq[t + 256], q1);
}

__global__ void bnparams_kernel(const float* __restrict__ sum, const float* __restrict__ sumsq,
                                const float* __restrict__ g, const float* __restrict__ be,
                                float* __restrict__ scale, float* __restrict__ shift, float invM) {
    const int c = blockIdx.x * blockDim.x + threadIdx.x;
    const float mu  = sum[c] * invM;
    const float var = sumsq[c] * invM - mu * mu;
    const float sc  = g[c] * rsqrtf(var + 1e-5f);
    scale[c] = sc;
    shift[c] = be[c] - mu * sc;
}

// ---------------- BN scale/shift + ReLU, bf16 output (feeds GEMM2) ----------------
__global__ void bnrelu_bf_kernel(const float* __restrict__ Y, const float* __restrict__ scale,
                                 const float* __restrict__ shift, __bf16* __restrict__ Xout) {
    const size_t i = (size_t)blockIdx.x * blockDim.x + threadIdx.x;
    const int c = (int)(i & (H_ - 1));
    const float v = Y[i] * scale[c] + shift[c];
    Xout[i] = (__bf16)fmaxf(v, 0.0f);
}

// ---------------- BN scale/shift + ReLU, fp32 in-place (final output) ----------------
__global__ void bnrelu_f32_kernel(float* __restrict__ Y, const float* __restrict__ scale,
                                  const float* __restrict__ shift) {
    const size_t i = (size_t)blockIdx.x * blockDim.x + threadIdx.x;
    const int c = (int)(i & (H_ - 1));
    const float v = Y[i] * scale[c] + shift[c];
    Y[i] = fmaxf(v, 0.0f);
}

extern "C" void kernel_launch(void* const* d_in, const int* in_sizes, int n_in,
                              void* d_out, int out_size, void* d_ws, size_t ws_size,
                              hipStream_t stream) {
    const float* xyz1    = (const float*)d_in[0];
    const float* xyz2    = (const float*)d_in[1];
    const float* points1 = (const float*)d_in[2];
    const float* points2 = (const float*)d_in[3];
    const float* W1      = (const float*)d_in[4];
    const float* b1      = (const float*)d_in[5];
    const float* g1      = (const float*)d_in[6];
    const float* be1     = (const float*)d_in[7];
    const float* W2      = (const float*)d_in[8];
    const float* b2      = (const float*)d_in[9];
    const float* g2      = (const float*)d_in[10];
    const float* be2     = (const float*)d_in[11];

    const size_t rows = (size_t)B_ * N_;   // 65536
    char* ws = (char*)d_ws;
    size_t off = 0;
    auto take = [&](size_t bytes) -> char* {
        char* p = ws + off;
        off = (off + bytes + 255) & ~(size_t)255;
        return p;
    };
    int*    idx   = (int*)take(rows * 3 * sizeof(int));
    float*  wgt   = (float*)take(rows * 3 * sizeof(float));
    __bf16* Xbf   = (__bf16*)take(rows * CIN_ * sizeof(__bf16));
    __bf16* W1bf  = (__bf16*)take((size_t)H_ * CIN_ * sizeof(__bf16));
    __bf16* W2bf  = (__bf16*)take((size_t)H_ * H_ * sizeof(__bf16));
    float*  Y1    = (float*)take(rows * H_ * sizeof(float));
    __bf16* X2bf  = (__bf16*)take(rows * H_ * sizeof(__bf16));
    float*  sum   = (float*)take(H_ * sizeof(float));
    float*  sumsq = (float*)take(H_ * sizeof(float));
    float*  scale = (float*)take(H_ * sizeof(float));
    float*  shift = (float*)take(H_ * sizeof(float));
    float*  Yout  = (float*)d_out;

    const float invM = 1.0f / (float)rows;

    // Stage 0: neighbor search + feature build (bf16 activations)
    knn3_kernel<<<dim3(N_ / 256, B_), 256, 0, stream>>>(xyz1, xyz2, idx, wgt);
    f2bf_kernel<<<(H_ * CIN_) / 256, 256, 0, stream>>>(W1, W1bf, H_ * CIN_);
    f2bf_kernel<<<(H_ * H_) / 256, 256, 0, stream>>>(W2, W2bf, H_ * H_);
    interp_concat_kernel<<<(unsigned)rows, 256, 0, stream>>>(points1, points2, idx, wgt, Xbf);

    // Stage 1: GEMM1 (K=384) + bias, then BN stats, then scale/shift+ReLU -> bf16
    gemm_bf16_kernel<CIN_><<<dim3((unsigned)(rows / 128), H_ / 64), 256, 0, stream>>>(Xbf, W1bf, b1, Y1);
    zero_kernel<<<2, 256, 0, stream>>>(sum, H_);
    zero_kernel<<<2, 256, 0, stream>>>(sumsq, H_);
    colstats_kernel<<<(unsigned)(rows / CHUNK_ROWS), 256, 0, stream>>>(Y1, sum, sumsq);
    bnparams_kernel<<<2, 256, 0, stream>>>(sum, sumsq, g1, be1, scale, shift, invM);
    bnrelu_bf_kernel<<<(unsigned)((rows * H_) / 256), 256, 0, stream>>>(Y1, scale, shift, X2bf);

    // Stage 2: GEMM2 (K=512) + bias -> d_out, BN stats, scale/shift+ReLU in place
    gemm_bf16_kernel<H_><<<dim3((unsigned)(rows / 128), H_ / 64), 256, 0, stream>>>(X2bf, W2bf, b2, Yout);
    zero_kernel<<<2, 256, 0, stream>>>(sum, H_);
    zero_kernel<<<2, 256, 0, stream>>>(sumsq, H_);
    colstats_kernel<<<(unsigned)(rows / CHUNK_ROWS), 256, 0, stream>>>(Yout, sum, sumsq);
    bnparams_kernel<<<2, 256, 0, stream>>>(sum, sumsq, g2, be2, scale, shift, invM);
    bnrelu_f32_kernel<<<(unsigned)((rows * H_) / 256), 256, 0, stream>>>(Yout, scale, shift);
}
